// SolubilityGNN_17497696764251
// MI455X (gfx1250) — compile-verified, matrix operand order checked by reading, not run on hardware
//
#include <hip/hip_runtime.h>
#include <hip/hip_bf16.h>
#include <math.h>
#include <float.h>

// ---------------------------------------------------------------------------
// SolubilityGNN for MI455X (gfx1250, wave32).
// Heavy GEMMs (layer-2 TransformerConv qkvs fused, GIN, FC1-3) use
// v_wmma_f32_16x16x32_f16 with pre-converted f16 operands:
//   A: row-major f16, per-lane 16 elements = two contiguous 16B chunks
//   B: col-major (transposed) f16, same two-chunk pattern
// so the inner loop is pure global_load_b128 + v_wmma, no branches.
// Edge softmax / aggregation use float atomics with edge-feature recompute.
// ---------------------------------------------------------------------------

#define NNODES 100000
#define NEDGES 300000
#define NGRAPH 4096
#define NHEAD  4
#define CHID   64
#define CDIM   256      // NHEAD * CHID
#define NDESC  200
#define HYBW   656      // 2*128 + 2*200
#define HYBP   672      // HYBW padded to multiple of 32
#define BN_EPS 1e-5f

typedef __attribute__((ext_vector_type(16))) _Float16 v16h;
typedef __attribute__((ext_vector_type(8)))  _Float16 v8h;
typedef __attribute__((ext_vector_type(8)))  float    v8f;

__device__ __forceinline__ float gelu_exact(float x) {
  return 0.5f * x * (1.0f + erff(x * 0.70710678118654752f));
}

__device__ __forceinline__ void atomicMaxF(float* addr, float val) {
  int* ai = (int*)addr;
  int old = __float_as_int(*addr);
  while (__int_as_float(old) < val) {
    int assumed = old;
    old = atomicCAS(ai, assumed, __float_as_int(val));
    if (old == assumed) break;
  }
}

// ---------------------------------------------------------------- utilities
__global__ void k_fill(float* p, long n, float v) {
  long i = (long)blockIdx.x * blockDim.x + threadIdx.x;
  if (i < n) p[i] = v;
}

__global__ void k_add(const float* a, const float* b, float* y, long n) {
  long i = (long)blockIdx.x * blockDim.x + threadIdx.x;
  if (i < n) y[i] = a[i] + b[i];
}

// f32 row-major [rows x Kin]  ->  f16 row-major [rows x Kout], zero-padded
__global__ void k_cvt_x(const float* __restrict__ x, _Float16* __restrict__ xh,
                        long rows, int Kin, int Kout) {
  long i = (long)blockIdx.x * blockDim.x + threadIdx.x;
  if (i >= rows * Kout) return;
  long r = i / Kout;
  int  k = (int)(i - r * Kout);
  xh[i] = (k < Kin) ? (_Float16)x[r * Kin + k] : (_Float16)0.f;
}

// f32 [Kin x Ncol] -> f16 transposed [Ncol x Kpad] (col-major, zero-padded K)
__global__ void k_cvt_wt(const float* __restrict__ w, _Float16* __restrict__ wt,
                         int Kin, int Kpad, int Ncol) {
  int i = blockIdx.x * blockDim.x + threadIdx.x;
  if (i >= Ncol * Kpad) return;
  int n = i / Kpad;
  int k = i - n * Kpad;
  wt[i] = (k < Kin) ? (_Float16)w[(long)k * Ncol + n] : (_Float16)0.f;
}

// ------------------------------------------------ layer-1 linears (din = 9)
__global__ void k_lin1(const float* __restrict__ x,
                       const float* Wq, const float* bq,
                       const float* Wk, const float* bk,
                       const float* Wv, const float* bv,
                       const float* Ws, const float* bs,
                       float* q, float* k, float* v, float* t) {
  int i = blockIdx.x * blockDim.x + threadIdx.x;
  if (i >= NNODES * CDIM) return;
  int n = i >> 8, c = i & 255;
  float xr[9];
#pragma unroll
  for (int j = 0; j < 9; ++j) xr[j] = x[n * 9 + j];
  float aq = bq[c], ak = bk[c], av = bv[c], as = bs[c];
#pragma unroll
  for (int j = 0; j < 9; ++j) {
    float xv = xr[j];
    aq += xv * Wq[j * CDIM + c];
    ak += xv * Wk[j * CDIM + c];
    av += xv * Wv[j * CDIM + c];
    as += xv * Ws[j * CDIM + c];
  }
  q[i] = aq; k[i] = ak; v[i] = av; t[i] = as;
}

// ------------------------------------------------------- edge attention ops
__global__ void k_alpha(const int* __restrict__ ei, const float* __restrict__ ea,
                        const float* __restrict__ We, const float* __restrict__ be,
                        const float* __restrict__ q, const float* __restrict__ k,
                        float* __restrict__ alpha, float* __restrict__ m) {
  int i = blockIdx.x * blockDim.x + threadIdx.x;
  if (i >= NEDGES * NHEAD) return;
  int e = i >> 2, h = i & 3;
  int src = ei[e], dst = ei[NEDGES + e];
  float er[6];
#pragma unroll
  for (int j = 0; j < 6; ++j) er[j] = ea[e * 6 + j];
  float acc = 0.f;
  int base = h * CHID;
  for (int c = 0; c < CHID; ++c) {
    int col = base + c;
    float ev = be[col];
#pragma unroll
    for (int j = 0; j < 6; ++j) ev += er[j] * We[j * CDIM + col];
    acc += q[(long)dst * CDIM + col] * (k[(long)src * CDIM + col] + ev);
  }
  acc *= 0.125f;                 // 1/sqrt(64)
  alpha[i] = acc;
  atomicMaxF(&m[dst * NHEAD + h], acc);
}

__global__ void k_expnorm(const int* __restrict__ ei, float* __restrict__ alpha,
                          const float* __restrict__ m, float* __restrict__ den) {
  int i = blockIdx.x * blockDim.x + threadIdx.x;
  if (i >= NEDGES * NHEAD) return;
  int e = i >> 2, h = i & 3;
  int dst = ei[NEDGES + e];
  float a = expf(alpha[i] - m[dst * NHEAD + h]);
  alpha[i] = a;
  atomicAdd(&den[dst * NHEAD + h], a);
}

__global__ void k_msg(const int* __restrict__ ei, const float* __restrict__ ea,
                      const float* __restrict__ We, const float* __restrict__ be,
                      const float* __restrict__ v, const float* __restrict__ alpha,
                      const float* __restrict__ den, float* __restrict__ t) {
  long i = (long)blockIdx.x * blockDim.x + threadIdx.x;
  if (i >= (long)NEDGES * CDIM) return;
  int e = (int)(i >> 8), col = (int)(i & 255);
  int h = col >> 6;
  int src = ei[e], dst = ei[NEDGES + e];
  float a = alpha[e * NHEAD + h] / (den[dst * NHEAD + h] + 1e-16f);
  float ev = be[col];
#pragma unroll
  for (int j = 0; j < 6; ++j) ev += ea[e * 6 + j] * We[j * CDIM + col];
  atomicAdd(&t[(long)dst * CDIM + col], (v[(long)src * CDIM + col] + ev) * a);
}

// -------------------------------------------------------------- batch norm
__global__ void k_bnstat(const float* __restrict__ x, float* __restrict__ stat,
                         int C, int rows) {
  int c = threadIdx.x;                  // blockDim.x == C
  float s = 0.f, ss = 0.f;
  for (int r = blockIdx.x; r < rows; r += gridDim.x) {
    float v = x[(long)r * C + c];
    s += v; ss += v * v;
  }
  atomicAdd(&stat[c], s);
  atomicAdd(&stat[C + c], ss);
}

__global__ void k_bnact(const float* __restrict__ x, const float* __restrict__ stat,
                        const float* __restrict__ g, const float* __restrict__ b,
                        float* __restrict__ y, int C, int rows) {
  long i = (long)blockIdx.x * blockDim.x + threadIdx.x;
  if (i >= (long)rows * C) return;
  int c = (int)(i % C);
  float inv = 1.f / (float)rows;
  float mu  = stat[c] * inv;
  float var = stat[C + c] * inv - mu * mu;
  float v = (x[i] - mu) * rsqrtf(var + BN_EPS) * g[c] + b[c];
  y[i] = gelu_exact(v);
}

// ---------------------------------------------------------------- GIN aggr
__global__ void k_ginagg(const int* __restrict__ ei, const float* __restrict__ h,
                         float* __restrict__ agg) {
  long i = (long)blockIdx.x * blockDim.x + threadIdx.x;
  if (i >= (long)NEDGES * CDIM) return;
  int e = (int)(i >> 8), c = (int)(i & 255);
  atomicAdd(&agg[(long)ei[NEDGES + e] * CDIM + c], h[(long)ei[e] * CDIM + c]);
}

// ------------------------------------------------------------------ pooling
__global__ void k_count(const int* __restrict__ batch, float* __restrict__ cnt) {
  int n = blockIdx.x * blockDim.x + threadIdx.x;
  if (n >= NNODES) return;
  atomicAdd(&cnt[batch[n]], 1.f);
}

__global__ void k_pool(const int* __restrict__ batch, const float* __restrict__ h,
                       float* __restrict__ pmax, float* __restrict__ psum) {
  int i = blockIdx.x * blockDim.x + threadIdx.x;
  if (i >= NNODES * CHID) return;
  int n = i >> 6, c = i & 63;
  int b = batch[n];
  float v = h[i];
  atomicMaxF(&pmax[b * CHID + c], v);
  atomicAdd(&psum[b * CHID + c], v);
}

__global__ void k_poolfin(const float* __restrict__ pmax, const float* __restrict__ psum,
                          const float* __restrict__ cnt, float* __restrict__ hyb, int off) {
  int i = blockIdx.x * blockDim.x + threadIdx.x;
  if (i >= NGRAPH * CHID) return;
  int g = i >> 6, c = i & 63;
  float ct = cnt[g];
  float mx = ct > 0.f ? pmax[i] : 0.f;
  float mn = psum[i] / fmaxf(ct, 1.f);
  hyb[g * HYBW + off + c] = mx;
  hyb[g * HYBW + off + CHID + c] = mn;
}

__global__ void k_desc(const float* __restrict__ d, float* __restrict__ hyb, int off) {
  int i = blockIdx.x * blockDim.x + threadIdx.x;
  if (i >= NGRAPH * NDESC) return;
  int g = i / NDESC, c = i - g * NDESC;
  hyb[g * HYBW + off + c] = d[i];
}

// ------------------------------------------------------------- final linear
__global__ void k_out(const float* __restrict__ z, const float* __restrict__ W,
                      const float* __restrict__ b, float* __restrict__ out) {
  int g = blockIdx.x * blockDim.x + threadIdx.x;
  if (g >= NGRAPH) return;
  float acc = b[0];
#pragma unroll
  for (int j = 0; j < 32; ++j) acc += z[g * 32 + j] * W[j];
  out[g] = acc;
}

// ------------------------------------------------------------- WMMA GEMM
// Y[w] = Xh(MxK f16) @ Wt[w](col-major NcolxK f16) + B[w], optional GELU.
// K must be a multiple of 32. Each wave owns one 16x16 D tile; NW weight
// matrices (packed consecutively in Wt, each Ncol*K halfs) share the A tile.
// Per CDNA5 ISA §7.12.2, lane 'l' (half = l>>4, l15 = l&15) holds:
//   A row  = tm*16+l15, elements e<8 -> K = k0+8*half+e,
//                       elements e>=8 -> K = k0+16+8*half+(e-8)
// i.e. two contiguous 16-byte chunks -> two global_load_b128 per matrix.
template <int NW>
struct GArgs { const float* B[NW]; float* Y[NW]; };

template <int NW>
__global__ void k_wmma_gemm(const _Float16* __restrict__ Xh,
                            const _Float16* __restrict__ Wt,
                            GArgs<NW> g, int M, int K, int Ncol, int act) {
  const int lane = threadIdx.x & 31;
  const int wave = blockIdx.x * (blockDim.x >> 5) + (threadIdx.x >> 5);
  const int tilesN = Ncol >> 4;
  const int tilesM = M >> 4;
  if (wave >= tilesM * tilesN) return;
  const int tm = wave / tilesN;
  const int tn = wave - tm * tilesN;
  const int half = lane >> 4;
  const int l15  = lane & 15;
  const int row = tm * 16 + l15;     // A row held by this lane
  const int col = tn * 16 + l15;     // B column / D column for this lane

  const _Float16* __restrict__ ap = Xh + (long)row * K + 8 * half;
  const _Float16* __restrict__ bp[NW];
#pragma unroll
  for (int w = 0; w < NW; ++w)
    bp[w] = Wt + (size_t)w * Ncol * K + (long)col * K + 8 * half;

  v8f acc[NW];
#pragma unroll
  for (int w = 0; w < NW; ++w)
#pragma unroll
    for (int r = 0; r < 8; ++r) acc[w][r] = 0.f;

#pragma unroll 2
  for (int k0 = 0; k0 < K; k0 += 32) {
    v8h a0 = *(const v8h*)(ap + k0);
    v8h a1 = *(const v8h*)(ap + k0 + 16);
    v16h a;
#pragma unroll
    for (int e = 0; e < 8; ++e) { a[e] = a0[e]; a[8 + e] = a1[e]; }
#pragma unroll
    for (int w = 0; w < NW; ++w) {
      v8h b0 = *(const v8h*)(bp[w] + k0);
      v8h b1 = *(const v8h*)(bp[w] + k0 + 16);
      v16h bm;
#pragma unroll
      for (int e = 0; e < 8; ++e) { bm[e] = b0[e]; bm[8 + e] = b1[e]; }
      acc[w] = __builtin_amdgcn_wmma_f32_16x16x32_f16(
          false, a, false, bm, (short)0, acc[w], false, false);
    }
  }
#pragma unroll
  for (int w = 0; w < NW; ++w) {
    float bias = g.B[w][col];
#pragma unroll
    for (int r = 0; r < 8; ++r) {
      int mrow = tm * 16 + r + 8 * half;
      float v = acc[w][r] + bias;
      if (act) v = gelu_exact(v);
      g.Y[w][(long)mrow * Ncol + col] = v;
    }
  }
}

// ---------------------------------------------------------------------------
extern "C" void kernel_launch(void* const* d_in, const int* in_sizes, int n_in,
                              void* d_out, int out_size, void* d_ws, size_t ws_size,
                              hipStream_t stream) {
  (void)in_sizes; (void)n_in; (void)out_size; (void)ws_size;
  // params flattened jax-pytree style (sorted keys, nested t1/t2 last):
  const float* bn1_b = (const float*)d_in[0];
  const float* bn1_g = (const float*)d_in[1];
  const float* bn2_b = (const float*)d_in[2];
  const float* bn2_g = (const float*)d_in[3];
  const float* bn3_b = (const float*)d_in[4];
  const float* bn3_g = (const float*)d_in[5];
  const float* fc1_W = (const float*)d_in[6];
  const float* fc1_b = (const float*)d_in[7];
  const float* fc2_W = (const float*)d_in[8];
  const float* fc2_b = (const float*)d_in[9];
  const float* fc3_W = (const float*)d_in[10];
  const float* fc3_b = (const float*)d_in[11];
  const float* gin_W = (const float*)d_in[12];
  const float* gin_b = (const float*)d_in[13];
  const float* out_W = (const float*)d_in[14];
  const float* out_b = (const float*)d_in[15];
  const float* t1_We = (const float*)d_in[16];
  const float* t1_Wk = (const float*)d_in[17];
  const float* t1_Wq = (const float*)d_in[18];
  const float* t1_Ws = (const float*)d_in[19];
  const float* t1_Wv = (const float*)d_in[20];
  const float* t1_be = (const float*)d_in[21];
  const float* t1_bk = (const float*)d_in[22];
  const float* t1_bq = (const float*)d_in[23];
  const float* t1_bs = (const float*)d_in[24];
  const float* t1_bv = (const float*)d_in[25];
  const float* t2_We = (const float*)d_in[26];
  const float* t2_Wk = (const float*)d_in[27];
  const float* t2_Wq = (const float*)d_in[28];
  const float* t2_Ws = (const float*)d_in[29];
  const float* t2_Wv = (const float*)d_in[30];
  const float* t2_be = (const float*)d_in[31];
  const float* t2_bk = (const float*)d_in[32];
  const float* t2_bq = (const float*)d_in[33];
  const float* t2_bs = (const float*)d_in[34];
  const float* t2_bv = (const float*)d_in[35];

  // workspace layout (floats, then f16 region); ~610 MB total
  float* ws = (float*)d_ws;
  const size_t NC = (size_t)NNODES * CDIM;
  float* Hb  = ws;            // N x 256 current features
  float* Qb  = ws + 1 * NC;   // N x 256 (also GIN agg)
  float* Kb  = ws + 2 * NC;
  float* Vb  = ws + 3 * NC;   // also GIN linear output (N x 64 prefix)
  float* Tb  = ws + 4 * NC;   // tconv out (skip + messages)
  float* Ab  = ws + 5 * NC;                         // E x 4 alpha
  float* Mb  = Ab + (size_t)NEDGES * NHEAD;         // N x 4 seg max
  float* Db  = Mb + (size_t)NNODES * NHEAD;         // N x 4 seg sum
  float* Sb  = Db + (size_t)NNODES * NHEAD;         // 2 x 256 BN stats
  float* Gb  = Sb + 2 * CDIM;                       // N x 64 GIN features
  float* PMb = Gb + (size_t)NNODES * CHID;          // G x 64 pool max
  float* PSb = PMb + (size_t)NGRAPH * CHID;         // G x 64 pool sum
  float* CTb = PSb + (size_t)NGRAPH * CHID;         // G counts
  float* Z1  = CTb + NGRAPH;                        // G x 128
  float* Z2  = Z1 + (size_t)NGRAPH * 128;           // G x 64
  float* Z3  = Z2 + (size_t)NGRAPH * 64;            // G x 32
  size_t fend = (size_t)(Z3 + (size_t)NGRAPH * 32 - ws);
  fend = (fend + 7) & ~(size_t)7;                   // 16B align the f16 region
  _Float16* Xh = (_Float16*)(ws + fend);            // N x 256 f16 (GEMM A)
  _Float16* Wh = Xh + NC;                           // up to 4 x 256 x 256 f16 (GEMM B, transposed)

  float* outp = (float*)d_out;         // G x 1
  float* hyb  = outp + NGRAPH;         // G x 656 (second tuple output)

  auto cdiv = [](long a, long b) { return (int)((a + b - 1) / b); };
  const int EH   = NEDGES * NHEAD;
  const long EC  = (long)NEDGES * CDIM;

  for (int s = 0; s < 2; ++s) {
    const float* x   = (const float*)d_in[36 + s * 5];
    const int*   ei  = (const int*)  d_in[37 + s * 5];
    const int*   bat = (const int*)  d_in[38 + s * 5];
    const float* dsc = (const float*)d_in[39 + s * 5];
    const float* eat = (const float*)d_in[40 + s * 5];

    auto run_edges = [&](const float* We, const float* be) {
      k_fill<<<cdiv(NNODES * NHEAD, 256), 256, 0, stream>>>(Mb, (long)NNODES * NHEAD, -FLT_MAX);
      k_fill<<<cdiv(NNODES * NHEAD, 256), 256, 0, stream>>>(Db, (long)NNODES * NHEAD, 0.f);
      k_alpha<<<cdiv(EH, 256), 256, 0, stream>>>(ei, eat, We, be, Qb, Kb, Ab, Mb);
      k_expnorm<<<cdiv(EH, 256), 256, 0, stream>>>(ei, Ab, Mb, Db);
      k_msg<<<cdiv(EC, 256), 256, 0, stream>>>(ei, eat, We, be, Vb, Ab, Db, Tb);
    };
    auto run_bn = [&](const float* xin, float* yout, const float* gg, const float* bb, int C) {
      k_fill<<<cdiv(2 * C, 256), 256, 0, stream>>>(Sb, 2 * C, 0.f);
      k_bnstat<<<1024, C, 0, stream>>>(xin, Sb, C, NNODES);
      k_bnact<<<cdiv((long)NNODES * C, 256), 256, 0, stream>>>(xin, Sb, gg, bb, yout, C, NNODES);
    };

    // ---- TransformerConv layer 1 (din=9, scalar linears; skip lands in Tb)
    k_lin1<<<cdiv((long)NNODES * CDIM, 256), 256, 0, stream>>>(
        x, t1_Wq, t1_bq, t1_Wk, t1_bk, t1_Wv, t1_bv, t1_Ws, t1_bs, Qb, Kb, Vb, Tb);
    run_edges(t1_We, t1_be);
    run_bn(Tb, Hb, bn1_g, bn1_b, CDIM);

    // ---- TransformerConv layer 2: fused q/k/v/skip WMMA GEMM, K=N=256
    {
      k_cvt_x<<<cdiv((long)NC, 256), 256, 0, stream>>>(Hb, Xh, NNODES, CDIM, CDIM);
      const float* Wsrc[4] = { t2_Wq, t2_Wk, t2_Wv, t2_Ws };
      for (int w = 0; w < 4; ++w)
        k_cvt_wt<<<cdiv(CDIM * CDIM, 256), 256, 0, stream>>>(
            Wsrc[w], Wh + (size_t)w * CDIM * CDIM, CDIM, CDIM, CDIM);
      GArgs<4> ga;
      ga.B[0] = t2_bq; ga.B[1] = t2_bk; ga.B[2] = t2_bv; ga.B[3] = t2_bs;
      ga.Y[0] = Qb;    ga.Y[1] = Kb;    ga.Y[2] = Vb;    ga.Y[3] = Tb;
      int tiles = (NNODES / 16) * (CDIM / 16);
      k_wmma_gemm<4><<<cdiv(tiles, 8), 256, 0, stream>>>(Xh, Wh, ga, NNODES, CDIM, CDIM, 0);
    }
    run_edges(t2_We, t2_be);
    run_bn(Tb, Hb, bn2_g, bn2_b, CDIM);

    // ---- GIN: agg + (h+agg) @ gin_W (WMMA, 256->64) + BN + GELU
    k_fill<<<cdiv((long)NC, 256), 256, 0, stream>>>(Qb, (long)NC, 0.f);
    k_ginagg<<<cdiv(EC, 256), 256, 0, stream>>>(ei, Hb, Qb);
    k_add<<<cdiv((long)NC, 256), 256, 0, stream>>>(Hb, Qb, Tb, (long)NC);
    {
      k_cvt_x<<<cdiv((long)NC, 256), 256, 0, stream>>>(Tb, Xh, NNODES, CDIM, CDIM);
      k_cvt_wt<<<cdiv(CHID * CDIM, 256), 256, 0, stream>>>(gin_W, Wh, CDIM, CDIM, CHID);
      GArgs<1> g1; g1.B[0] = gin_b; g1.Y[0] = Vb;
      int tiles = (NNODES / 16) * (CHID / 16);
      k_wmma_gemm<1><<<cdiv(tiles, 8), 256, 0, stream>>>(Xh, Wh, g1, NNODES, CDIM, CHID, 0);
    }
    run_bn(Vb, Gb, bn3_g, bn3_b, CHID);

    // ---- global max+mean pool, write into hybrid slot
    k_fill<<<cdiv(NGRAPH * CHID, 256), 256, 0, stream>>>(PMb, (long)NGRAPH * CHID, -FLT_MAX);
    k_fill<<<cdiv(NGRAPH * CHID, 256), 256, 0, stream>>>(PSb, (long)NGRAPH * CHID, 0.f);
    k_fill<<<cdiv(NGRAPH, 256), 256, 0, stream>>>(CTb, (long)NGRAPH, 0.f);
    k_count<<<cdiv(NNODES, 256), 256, 0, stream>>>(bat, CTb);
    k_pool<<<cdiv(NNODES * CHID, 256), 256, 0, stream>>>(bat, Gb, PMb, PSb);
    k_poolfin<<<cdiv(NGRAPH * CHID, 256), 256, 0, stream>>>(PMb, PSb, CTb, hyb, s * 128);
    k_desc<<<cdiv(NGRAPH * NDESC, 256), 256, 0, stream>>>(dsc, hyb, 256 + s * NDESC);
  }

  // ---- MLP head (WMMA GEMMs with fused exact GELU), then final 32->1
  {
    // FC1: K = 656 padded to 672
    k_cvt_x<<<cdiv((long)NGRAPH * HYBP, 256), 256, 0, stream>>>(hyb, Xh, NGRAPH, HYBW, HYBP);
    k_cvt_wt<<<cdiv(128 * HYBP, 256), 256, 0, stream>>>(fc1_W, Wh, HYBW, HYBP, 128);
    GArgs<1> f1; f1.B[0] = fc1_b; f1.Y[0] = Z1;
    k_wmma_gemm<1><<<cdiv((NGRAPH / 16) * (128 / 16), 8), 256, 0, stream>>>(
        Xh, Wh, f1, NGRAPH, HYBP, 128, 1);
    // FC2: 128 -> 64
    k_cvt_x<<<cdiv((long)NGRAPH * 128, 256), 256, 0, stream>>>(Z1, Xh, NGRAPH, 128, 128);
    k_cvt_wt<<<cdiv(64 * 128, 256), 256, 0, stream>>>(fc2_W, Wh, 128, 128, 64);
    GArgs<1> f2; f2.B[0] = fc2_b; f2.Y[0] = Z2;
    k_wmma_gemm<1><<<cdiv((NGRAPH / 16) * (64 / 16), 8), 256, 0, stream>>>(
        Xh, Wh, f2, NGRAPH, 128, 64, 1);
    // FC3: 64 -> 32
    k_cvt_x<<<cdiv((long)NGRAPH * 64, 256), 256, 0, stream>>>(Z2, Xh, NGRAPH, 64, 64);
    k_cvt_wt<<<cdiv(32 * 64, 256), 256, 0, stream>>>(fc3_W, Wh, 64, 64, 32);
    GArgs<1> f3; f3.B[0] = fc3_b; f3.Y[0] = Z3;
    k_wmma_gemm<1><<<cdiv((NGRAPH / 16) * (32 / 16), 8), 256, 0, stream>>>(
        Xh, Wh, f3, NGRAPH, 64, 32, 1);
  }
  k_out<<<cdiv(NGRAPH, 256), 256, 0, stream>>>(Z3, out_W, out_b, outp);
}
